// Transformer_76287209111671
// MI455X (gfx1250) — matrix operand order repaired and checked by reading, not compiled, not run on hardware
//
#include <hip/hip_runtime.h>
#include <stdint.h>

typedef __attribute__((ext_vector_type(16))) __bf16 v16bf;
typedef __attribute__((ext_vector_type(8)))  float  v8f;
typedef __attribute__((ext_vector_type(4)))  unsigned int uint4v;

#define LNUM 6
#define HNUM 8
#define MDIM 1024
#define HID 2048
#define BNUM 4
#define SNUM 1024
#define EDIM 128
#define EPS 1e-5f

static const size_t BSM  = (size_t)BNUM * SNUM * MDIM;        // 4194304
static const size_t BHSE = (size_t)BNUM * HNUM * SNUM * EDIM; // == BSM

// ---------------------------------------------------------------- elementwise

__global__ __launch_bounds__(256) void pos_encode_kernel(const float* __restrict__ x,
                                                         float* __restrict__ h,
                                                         __bf16* __restrict__ hbf) {
    size_t idx = (size_t)blockIdx.x * 256 + threadIdx.x;
    int m = (int)(idx & (MDIM - 1));
    int s = (int)((idx >> 10) & (SNUM - 1));
    float expo = (float)(m & ~1) / (float)MDIM;
    float div  = __expf(expo * 9.2103403719761836f);   // ln(10000)
    float ang  = (float)s / div;
    float pe   = ((m & 1) == 0) ? __sinf(ang) : __cosf(ang);
    float v = x[idx] + pe;
    h[idx] = v;
    hbf[idx] = (__bf16)v;
}

// Fused QKV weight in [N][K] layout: Wqkv[n][m] = w{q,k,v}[l][h][m][e], n = t*1024 + h*128 + e
__global__ __launch_bounds__(256) void convert_qkv_kernel(const float* __restrict__ wq,
                                                          const float* __restrict__ wk,
                                                          const float* __restrict__ wv,
                                                          __bf16* __restrict__ dst) {
    size_t idx = (size_t)blockIdx.x * 256 + threadIdx.x;   // n * M + m
    int m = (int)(idx & (MDIM - 1));
    int n = (int)(idx >> 10);
    int t = n >> 10;
    int hh = (n >> 7) & (HNUM - 1);
    int e = n & (EDIM - 1);
    const float* src = (t == 0) ? wq : (t == 1) ? wk : wv;
    dst[idx] = (__bf16)src[((size_t)hh * MDIM + m) * EDIM + e];
}

// plain fp32 -> bf16 cast (proj_w / w1_w / w2_w are already [N][K])
__global__ __launch_bounds__(256) void convert_cast_kernel(__bf16* __restrict__ dst,
                                                           const float* __restrict__ src) {
    size_t idx = (size_t)blockIdx.x * 256 + threadIdx.x;
    dst[idx] = (__bf16)src[idx];
}

// ---------------------------------------------------------------- GEMM (bf16 -> f32 acc)
// C[M x N] = alpha * A[M x K] * B[N x K]^T (+ bias) ; tiles 128x128x32, double-buffered.

struct GemmArgs {
    const __bf16* A;     // [Mrows][K] row-major
    const __bf16* B;     // [N][K] row-major
    float*  Cf;          // optional f32 out
    __bf16* Cb;          // optional bf16 out
    const float* bias;   // optional, indexed by global col
    long long aStrideZ;
    long long bStrideZ;
    long long cS1, cS2;  // c offset = (z/cDiv)*cS1 + (z%cDiv)*cS2
    int cDiv;
    int ldc;
    int lda, ldb;
    int K;
    int relu;
    int qkv;             // fused-QKV scatter mode (Cb only)
    float alpha;
};

__global__ __launch_bounds__(256) void gemm_bf16_kernel(GemmArgs p) {
    __shared__ __bf16 As[2][128][40];   // 80B row stride: 16B-aligned, 20-bank (conflict-free)
    __shared__ __bf16 Bs[2][128][40];

    const int tid  = threadIdx.x;
    const int z    = blockIdx.z;
    const int row0 = blockIdx.y * 128;
    const int n0   = blockIdx.x * 128;

    const __bf16* Aptr = p.A + (size_t)z * p.aStrideZ;
    const __bf16* Bptr = p.B + (size_t)z * p.bStrideZ;

    const int lane = tid & 31;
    const int wave = tid >> 5;
    const int wr = wave & 3, wc = wave >> 2;      // 4 x 2 wave grid; wave tile 32 x 64
    const int lrow = lane & 15, half = lane >> 4;

    const int ldr = (tid + 0 * 256) >> 2, ldc0 = ((tid) & 3) * 8;   // staging coords
    const int ldr1 = (tid + 256) >> 2;

    v8f acc[2][4] = {};
    uint4v ra[2], rb[2];

    const int nk = p.K >> 5;

    // ---- prologue: tile 0 -> regs -> LDS buf 0
    {
        ra[0] = *(const uint4v*)(Aptr + (size_t)(row0 + ldr)  * p.lda + ldc0);
        ra[1] = *(const uint4v*)(Aptr + (size_t)(row0 + ldr1) * p.lda + ldc0);
        rb[0] = *(const uint4v*)(Bptr + (size_t)(n0 + ldr)  * p.ldb + ldc0);
        rb[1] = *(const uint4v*)(Bptr + (size_t)(n0 + ldr1) * p.ldb + ldc0);
        *(uint4v*)&As[0][ldr][ldc0]  = ra[0];
        *(uint4v*)&As[0][ldr1][ldc0] = ra[1];
        *(uint4v*)&Bs[0][ldr][ldc0]  = rb[0];
        *(uint4v*)&Bs[0][ldr1][ldc0] = rb[1];
    }
    __syncthreads();

    for (int kt = 0; kt < nk; ++kt) {
        const int cur = kt & 1;
        const bool more = (kt + 1 < nk);
        if (more) {
            int k0 = (kt + 1) << 5;
            ra[0] = *(const uint4v*)(Aptr + (size_t)(row0 + ldr)  * p.lda + k0 + ldc0);
            ra[1] = *(const uint4v*)(Aptr + (size_t)(row0 + ldr1) * p.lda + k0 + ldc0);
            rb[0] = *(const uint4v*)(Bptr + (size_t)(n0 + ldr)  * p.ldb + k0 + ldc0);
            rb[1] = *(const uint4v*)(Bptr + (size_t)(n0 + ldr1) * p.ldb + k0 + ldc0);
        }

        union Frag { uint4v q[2]; v16bf v; };
        Frag af[2], bfr[4];
        #pragma unroll
        for (int i = 0; i < 2; ++i) {
            int r = wr * 32 + i * 16 + lrow;
            af[i].q[0] = *(const uint4v*)&As[cur][r][half * 8];
            af[i].q[1] = *(const uint4v*)&As[cur][r][16 + half * 8];
        }
        #pragma unroll
        for (int j = 0; j < 4; ++j) {
            int n = wc * 64 + j * 16 + lrow;
            bfr[j].q[0] = *(const uint4v*)&Bs[cur][n][half * 8];
            bfr[j].q[1] = *(const uint4v*)&Bs[cur][n][16 + half * 8];
        }

        #pragma unroll
        for (int i = 0; i < 2; ++i)
            #pragma unroll
            for (int j = 0; j < 4; ++j)
                acc[i][j] = __builtin_amdgcn_wmma_f32_16x16x32_bf16(
                    false, af[i].v, false, bfr[j].v, (short)0, acc[i][j], false, false);

        if (more) {
            int nxt = cur ^ 1;
            *(uint4v*)&As[nxt][ldr][ldc0]  = ra[0];
            *(uint4v*)&As[nxt][ldr1][ldc0] = ra[1];
            *(uint4v*)&Bs[nxt][ldr][ldc0]  = rb[0];
            *(uint4v*)&Bs[nxt][ldr1][ldc0] = rb[1];
        }
        __syncthreads();
    }

    // epilogue: C layout — VGPR r: lanes 0-15 -> (M=r, N=lane), lanes 16-31 -> (M=r+8, N=lane-16)
    size_t coff = (size_t)(z / p.cDiv) * p.cS1 + (size_t)(z % p.cDiv) * p.cS2;
    #pragma unroll
    for (int i = 0; i < 2; ++i) {
        #pragma unroll
        for (int j = 0; j < 4; ++j) {
            #pragma unroll
            for (int r = 0; r < 8; ++r) {
                int grow = row0 + wr * 32 + i * 16 + half * 8 + r;
                int gcol = n0 + wc * 64 + j * 16 + lrow;
                float v = acc[i][j][r] * p.alpha;
                if (p.bias) v += p.bias[gcol];
                if (p.relu) v = fmaxf(v, 0.0f);
                if (p.qkv) {
                    int t = gcol >> 10;
                    int hh = (gcol >> 7) & (HNUM - 1);
                    int e = gcol & (EDIM - 1);
                    int b = grow >> 10;
                    int s = grow & (SNUM - 1);
                    size_t dst;
                    if (t < 2)   // q, k: [b,h,s,e]
                        dst = (size_t)t * BHSE + (((size_t)b * HNUM + hh) * SNUM + s) * EDIM + e;
                    else         // v stored transposed: [b,h,e,s]
                        dst = 2 * BHSE + (((size_t)b * HNUM + hh) * EDIM + e) * SNUM + s;
                    p.Cb[dst] = (__bf16)v;
                } else {
                    size_t off = coff + (size_t)grow * p.ldc + gcol;
                    if (p.Cf) p.Cf[off] = v;
                    if (p.Cb) p.Cb[off] = (__bf16)v;
                }
            }
        }
    }
}

// ---------------------------------------------------------------- softmax (masked, per row)

__global__ __launch_bounds__(256) void softmax_kernel(const __bf16* __restrict__ scores,
                                                      __bf16* __restrict__ P,
                                                      const unsigned char* __restrict__ mask) {
    __shared__ float red[256];
    int z = blockIdx.y;                 // b*H + h
    int row = blockIdx.x;
    int b = z >> 3;                     // H == 8
    const __bf16* srow = scores + ((size_t)z * SNUM + row) * SNUM;
    __bf16* prow = P + ((size_t)z * SNUM + row) * SNUM;
    const unsigned char* mrow = mask + (size_t)b * SNUM;
    int tid = threadIdx.x;

    float vals[4];
    float lmax = -3.4e38f;
    #pragma unroll
    for (int i = 0; i < 4; ++i) {
        int c = tid + i * 256;
        float v = mrow[c] ? (float)srow[c] : -3.4e38f;
        vals[i] = v;
        lmax = fmaxf(lmax, v);
    }
    red[tid] = lmax; __syncthreads();
    for (int s = 128; s > 0; s >>= 1) {
        if (tid < s) red[tid] = fmaxf(red[tid], red[tid + s]);
        __syncthreads();
    }
    float smax = red[0]; __syncthreads();

    float lsum = 0.f;
    #pragma unroll
    for (int i = 0; i < 4; ++i) {
        int c = tid + i * 256;
        float e = mrow[c] ? __expf(vals[i] - smax) : 0.f;
        vals[i] = e;
        lsum += e;
    }
    red[tid] = lsum; __syncthreads();
    for (int s = 128; s > 0; s >>= 1) {
        if (tid < s) red[tid] += red[tid + s];
        __syncthreads();
    }
    float inv = 1.0f / red[0];
    #pragma unroll
    for (int i = 0; i < 4; ++i) {
        int c = tid + i * 256;
        prow[c] = (__bf16)(vals[i] * inv);
    }
}

// ---------------------------------------------------------------- residual + layernorm

__global__ __launch_bounds__(256) void add_ln_kernel(const float* __restrict__ a,
                                                     const float* __restrict__ res,
                                                     const float* __restrict__ g,
                                                     const float* __restrict__ bb,
                                                     float* __restrict__ hout,
                                                     __bf16* __restrict__ hbf,
                                                     float* __restrict__ extra) {
    __shared__ float red[256];
    size_t row = blockIdx.x;
    int tid = threadIdx.x;
    const float* arow = a + row * MDIM;
    const float* rrow = res + row * MDIM;

    float v[4];
    float lsum = 0.f;
    #pragma unroll
    for (int i = 0; i < 4; ++i) {
        int c = tid + i * 256;
        v[i] = arow[c] + rrow[c];
        lsum += v[i];
    }
    red[tid] = lsum; __syncthreads();
    for (int s = 128; s > 0; s >>= 1) {
        if (tid < s) red[tid] += red[tid + s];
        __syncthreads();
    }
    float mu = red[0] * (1.0f / MDIM); __syncthreads();

    float lvar = 0.f;
    #pragma unroll
    for (int i = 0; i < 4; ++i) {
        float d = v[i] - mu;
        lvar += d * d;
    }
    red[tid] = lvar; __syncthreads();
    for (int s = 128; s > 0; s >>= 1) {
        if (tid < s) red[tid] += red[tid + s];
        __syncthreads();
    }
    float rstd = rsqrtf(red[0] * (1.0f / MDIM) + EPS);

    #pragma unroll
    for (int i = 0; i < 4; ++i) {
        int c = tid + i * 256;
        float y = (v[i] - mu) * rstd * g[c] + bb[c];
        hout[row * MDIM + c] = y;
        hbf[row * MDIM + c] = (__bf16)y;
        if (extra) extra[row * MDIM + c] = y;
    }
}

// ---------------------------------------------------------------- host

extern "C" void kernel_launch(void* const* d_in, const int* in_sizes, int n_in,
                              void* d_out, int out_size, void* d_ws, size_t ws_size,
                              hipStream_t stream) {
    const float* x      = (const float*)d_in[0];
    const float* wq     = (const float*)d_in[1];
    const float* wk     = (const float*)d_in[2];
    const float* wv     = (const float*)d_in[3];
    const float* proj_w = (const float*)d_in[4];
    const float* ln1_g  = (const float*)d_in[5];
    const float* ln1_b  = (const float*)d_in[6];
    const float* w1_w   = (const float*)d_in[7];
    const float* w1_b   = (const float*)d_in[8];
    const float* w2_w   = (const float*)d_in[9];
    const float* w2_b   = (const float*)d_in[10];
    const float* ln2_g  = (const float*)d_in[11];
    const float* ln2_b  = (const float*)d_in[12];
    const unsigned char* mask = (const unsigned char*)d_in[13];
    float* out = (float*)d_out;

    char* ws = (char*)d_ws;
    size_t off = 0;
    auto alloc = [&](size_t bytes) -> void* {
        void* p = ws + off;
        off = (off + bytes + 255) & ~(size_t)255;
        return p;
    };

    float*  h      = (float*)alloc(BSM * 4);
    __bf16* h_bf   = (__bf16*)alloc(BSM * 2);
    __bf16* Wqkv   = (__bf16*)alloc((size_t)MDIM * 3 * MDIM * 2);
    __bf16* Wproj  = (__bf16*)alloc((size_t)MDIM * MDIM * 2);
    __bf16* W1     = (__bf16*)alloc((size_t)MDIM * HID * 2);
    __bf16* W2     = (__bf16*)alloc((size_t)HID * MDIM * 2);
    __bf16* qkvb   = (__bf16*)alloc(3 * BHSE * 2);           // q | k | v^T
    __bf16* scores = (__bf16*)alloc((size_t)BNUM * HNUM * SNUM * SNUM * 2);
    __bf16* Pmat   = (__bf16*)alloc((size_t)BNUM * HNUM * SNUM * SNUM * 2);
    __bf16* o_bf   = (__bf16*)alloc(BSM * 2);
    float*  attn   = (float*)alloc(BSM * 4);
    __bf16* mid    = (__bf16*)alloc((size_t)BNUM * SNUM * HID * 2);
    float*  ffn    = (float*)alloc(BSM * 4);

    dim3 blk(256);
    const float inv_scale = 0.08838834764831845f;   // 1/sqrt(128)

    pos_encode_kernel<<<(unsigned)(BSM / 256), blk, 0, stream>>>(x, h, h_bf);

    for (int l = 0; l < LNUM; ++l) {
        size_t wOff = (size_t)l * HNUM * MDIM * EDIM;
        convert_qkv_kernel<<<(unsigned)(3ull * MDIM * MDIM / 256), blk, 0, stream>>>(
            wq + wOff, wk + wOff, wv + wOff, Wqkv);
        convert_cast_kernel<<<(unsigned)((size_t)MDIM * MDIM / 256), blk, 0, stream>>>(
            Wproj, proj_w + (size_t)l * MDIM * MDIM);
        convert_cast_kernel<<<(unsigned)((size_t)HID * MDIM / 256), blk, 0, stream>>>(
            W1, w1_w + (size_t)l * HID * MDIM);
        convert_cast_kernel<<<(unsigned)((size_t)MDIM * HID / 256), blk, 0, stream>>>(
            W2, w2_w + (size_t)l * MDIM * HID);

        // fused QKV: (B*S x M) @ Wqkv[N=3M][K=M]^T -> scatter q/k/v^T bf16
        {
            GemmArgs g{};
            g.A = h_bf; g.B = Wqkv; g.Cb = qkvb;
            g.lda = MDIM; g.ldb = MDIM; g.K = MDIM;
            g.cDiv = 1; g.alpha = 1.0f; g.qkv = 1;
            gemm_bf16_kernel<<<dim3(3 * MDIM / 128, BNUM * SNUM / 128, 1), blk, 0, stream>>>(g);
        }
        // scores = Q @ K^T / sqrt(E) per (b,h):  A=Q[S][E], B=K[N=S][K=E]
        {
            GemmArgs g{};
            g.A = qkvb; g.B = qkvb + BHSE; g.Cb = scores;
            g.aStrideZ = (long long)SNUM * EDIM; g.bStrideZ = (long long)SNUM * EDIM;
            g.lda = EDIM; g.ldb = EDIM; g.K = EDIM;
            g.cDiv = 1; g.cS1 = (long long)SNUM * SNUM; g.ldc = SNUM;
            g.alpha = inv_scale;
            gemm_bf16_kernel<<<dim3(SNUM / 128, SNUM / 128, BNUM * HNUM), blk, 0, stream>>>(g);
        }
        softmax_kernel<<<dim3(SNUM, BNUM * HNUM), blk, 0, stream>>>(scores, Pmat, mask);
        // O = P @ V per (b,h):  A=P[S][S], B=V^T[N=E][K=S], scattered into (B,S,H*E)
        {
            GemmArgs g{};
            g.A = Pmat; g.B = qkvb + 2 * BHSE; g.Cb = o_bf;
            g.aStrideZ = (long long)SNUM * SNUM; g.bStrideZ = (long long)EDIM * SNUM;
            g.lda = SNUM; g.ldb = SNUM; g.K = SNUM;
            g.cDiv = HNUM; g.cS1 = (long long)SNUM * MDIM; g.cS2 = EDIM; g.ldc = MDIM;
            g.alpha = 1.0f;
            gemm_bf16_kernel<<<dim3(EDIM / 128, SNUM / 128, BNUM * HNUM), blk, 0, stream>>>(g);
        }
        // proj: (B*S x M) @ proj_w[N=M][K=M]^T -> attn f32
        {
            GemmArgs g{};
            g.A = o_bf; g.B = Wproj; g.Cf = attn;
            g.lda = MDIM; g.ldb = MDIM; g.K = MDIM;
            g.cDiv = 1; g.ldc = MDIM; g.alpha = 1.0f;
            gemm_bf16_kernel<<<dim3(MDIM / 128, BNUM * SNUM / 128, 1), blk, 0, stream>>>(g);
        }
        add_ln_kernel<<<(unsigned)(BNUM * SNUM), blk, 0, stream>>>(
            attn, h, ln1_g + (size_t)l * MDIM, ln1_b + (size_t)l * MDIM, h, h_bf, nullptr);
        // FFN1: relu(h @ W1^T + b1) -> mid bf16   (W1 = w1_w is [HID][M] = [N][K])
        {
            GemmArgs g{};
            g.A = h_bf; g.B = W1; g.Cb = mid; g.bias = w1_b + (size_t)l * HID;
            g.lda = MDIM; g.ldb = MDIM; g.K = MDIM;
            g.cDiv = 1; g.ldc = HID; g.alpha = 1.0f; g.relu = 1;
            gemm_bf16_kernel<<<dim3(HID / 128, BNUM * SNUM / 128, 1), blk, 0, stream>>>(g);
        }
        // FFN2: mid @ W2^T + b2 -> ffn f32   (W2 = w2_w is [M][HID] = [N][K])
        {
            GemmArgs g{};
            g.A = mid; g.B = W2; g.Cf = ffn; g.bias = w2_b + (size_t)l * MDIM;
            g.lda = HID; g.ldb = HID; g.K = HID;
            g.cDiv = 1; g.ldc = MDIM; g.alpha = 1.0f;
            gemm_bf16_kernel<<<dim3(MDIM / 128, BNUM * SNUM / 128, 1), blk, 0, stream>>>(g);
        }
        add_ln_kernel<<<(unsigned)(BNUM * SNUM), blk, 0, stream>>>(
            ffn, h, ln2_g + (size_t)l * MDIM, ln2_b + (size_t)l * MDIM, h, h_bf,
            out + (size_t)l * BSM);
    }
}
